// DeepModel3_29119878267295
// MI455X (gfx1250) — matrix operand order repaired and checked by reading, not compiled, not run on hardware
//
#include <hip/hip_runtime.h>
#include <hip/hip_bf16.h>
#include <stdint.h>

// ---------------- types ----------------
typedef __attribute__((ext_vector_type(8)))  _Float16 v8h;
typedef __attribute__((ext_vector_type(16))) _Float16 v16h;
typedef __attribute__((ext_vector_type(8)))  float    v8f;
typedef __attribute__((ext_vector_type(4)))  float    f32x4;
typedef __attribute__((ext_vector_type(4)))  unsigned int u32x4;
typedef __attribute__((ext_vector_type(8)))  int      i32x8;
typedef __attribute__((ext_vector_type(4)))  int      i32x4;

#define ASTRIDE 264   // activation LDS row stride in halves (256 data + 8 pad)
#define MTILE   256   // batch rows per block
#define NTHREADS 512  // 16 wave32

// ---------------- layer table (compile-time) ----------------
struct LD { int grp; int idx; int N; int K; int KP; unsigned woff; unsigned boff; };
struct Tab { LD l[48]; };

constexpr Tab mk_tab() {
  Tab t{};
  unsigned wo = 0, bo = 0;
  for (int li = 0; li < 47; ++li) {
    int grp = 0, idx = 0, N = 0, K = 0;
    if      (li == 0)  { grp = 0; idx = 0;      N = 256; K = 256; }
    else if (li <= 3)  { grp = 1; idx = li - 1; N = 256; K = 256; }
    else if (li == 4)  { grp = 2; idx = 0;      N = 64;  K = 256; }
    else if (li <= 25) { grp = 3; idx = li - 5; N = 64;  K = 64;  }
    else if (li == 26) { grp = 4; idx = 0;      N = 32;  K = 64;  }
    else if (li <= 35) { grp = 5; idx = li - 27;N = 32;  K = 32;  }
    else if (li == 36) { grp = 6; idx = 0;      N = 16;  K = 32;  }
    else               { grp = 7; idx = li - 37;N = 16;  K = 16;  }
    int KP = (K == 256) ? 264 : (K == 64) ? 72 : 40;   // K=32/16 -> 40 (K=16 zero-padded to 32)
    t.l[li] = LD{grp, idx, N, K, KP, wo, bo};
    wo += (unsigned)(N * KP);
    bo += (unsigned)N;
  }
  // layer 47: 16 -> 1, scalar dot. w47(16 floats) + b47(1) in bias region.
  t.l[47] = LD{8, 0, 1, 16, 0, wo, bo};
  return t;
}

constexpr Tab g_tab = mk_tab();
constexpr unsigned WTOT_HALVES = g_tab.l[47].woff;

// ---------------- TDM weight staging ----------------
#ifndef USE_TDM
#define USE_TDM 1
#endif
#if USE_TDM && !__has_builtin(__builtin_amdgcn_tensor_load_to_lds)
#undef USE_TDM
#define USE_TDM 0
#endif

__device__ __forceinline__ void wait_tensorcnt0() {
#if __has_builtin(__builtin_amdgcn_s_wait_tensorcnt)
  __builtin_amdgcn_s_wait_tensorcnt(0);
#else
  asm volatile("s_wait_tensorcnt 0" ::: "memory");
#endif
}

#if USE_TDM
// 2D tile: [N rows] x [KP halves], contiguous (stride == KP), global f16 -> LDS.
__device__ __forceinline__ void tdm_load_w(void* ldsDst, const _Float16* gsrc, int KP, int N) {
  u32x4 g0; i32x8 g1; i32x4 g2; i32x4 g3;
  for (int i = 0; i < 4; ++i) { g2[i] = 0; g3[i] = 0; }
  unsigned lds = (unsigned)(uintptr_t)ldsDst;
  unsigned long long ga = (unsigned long long)(uintptr_t)gsrc;

  g0[0] = 1u;                                                 // count=1 user descriptor
  g0[1] = lds;                                                // lds_addr
  g0[2] = (unsigned)(ga & 0xFFFFFFFFu);                       // global_addr lo
  g0[3] = (unsigned)((ga >> 32) & 0x1FFFFFFu) | (2u << 30);   // global_addr hi | type=2

  unsigned td0 = (unsigned)KP, td1 = (unsigned)N;
  g1[0] = (int)(1u << 16);                                    // wg_mask=0, data_size=1 (2B)
  g1[1] = (int)((td0 & 0xFFFFu) << 16);                       // tensor_dim0 lo16
  g1[2] = (int)(((td0 >> 16) & 0xFFFFu) | ((td1 & 0xFFFFu) << 16));
  g1[3] = (int)(((td1 >> 16) & 0xFFFFu) | ((unsigned)KP << 16)); // tile_dim0 = KP
  g1[4] = (int)(td1 & 0xFFFFu);                               // tile_dim1 = N, tile_dim2 = 0
  g1[5] = (int)KP;                                            // tensor_dim0_stride lo32
  g1[6] = 0;
  g1[7] = 0;
#if __clang_major__ >= 23
  i32x8 gpad; for (int i = 0; i < 8; ++i) gpad[i] = 0;
  __builtin_amdgcn_tensor_load_to_lds(g0, g1, g2, g3, gpad, 0);
#else
  __builtin_amdgcn_tensor_load_to_lds(g0, g1, g2, g3, 0);
#endif
}
#endif

__device__ __forceinline__ void stage_weights(_Float16* dst, const _Float16* src,
                                              int KP, int N, int tid) {
#if USE_TDM
  if (tid == 0) tdm_load_w(dst, src, KP, N);
#else
  int total = N * KP;                                         // multiple of 8
  for (int e = tid * 8; e < total; e += NTHREADS * 8)
    *(v8h*)(dst + e) = *(const v8h*)(src + e);
#endif
}

// ---------------- WMMA layer compute ----------------
// Each wave owns rows [rbase, rbase+16); accumulate all NT column tiles in registers.
template <int NT, int K, int KP>
__device__ __forceinline__ void layer_compute(_Float16* act, const _Float16* wsm,
                                              const float* bsm, int lane, int rbase,
                                              bool zpad16) {
  v8f zero = {0.f, 0.f, 0.f, 0.f, 0.f, 0.f, 0.f, 0.f};
  v8f acc[NT];
#pragma unroll
  for (int i = 0; i < NT; ++i) acc[i] = zero;

  const int l15   = lane & 15;
  const int hi    = lane >> 4;        // 0 or 1
  const _Float16* arow  = act + (rbase + l15) * ASTRIDE + hi * 8;   // A K-interleave
  const _Float16* wbase = wsm + l15 * KP + hi * 8;                  // B K-interleave

#pragma unroll 1
  for (int kk = 0; kk < K; kk += 32) {
    v8h alo = *(const v8h*)(arow + kk);
    v8h ahi = *(const v8h*)(arow + kk + 16);
    v16h a = __builtin_shufflevector(alo, ahi, 0,1,2,3,4,5,6,7,8,9,10,11,12,13,14,15);
#pragma unroll
    for (int nt = 0; nt < NT; ++nt) {
      const _Float16* wp = wbase + nt * (16 * KP) + kk;             // compile-time tile offset
      v8h blo = *(const v8h*)(wp);
      v8h bhi = *(const v8h*)(wp + 16);
      v16h b = __builtin_shufflevector(blo, bhi, 0,1,2,3,4,5,6,7,8,9,10,11,12,13,14,15);
      acc[nt] = __builtin_amdgcn_wmma_f32_16x16x32_f16(
          false, a, false, b, (short)0, acc[nt], false, false);
    }
  }

  // epilogue: bias + ReLU + f32->f16, scatter per C/D VGPR layout
#pragma unroll
  for (int nt = 0; nt < NT; ++nt) {
    int c = nt * 16 + l15;
    float bv = bsm[c];
#pragma unroll
    for (int v = 0; v < 8; ++v) {
      int m = rbase + v + 8 * hi;
      float r = acc[nt][v] + bv;
      r = r > 0.f ? r : 0.f;
      act[m * ASTRIDE + c] = (_Float16)r;
      if (zpad16) act[m * ASTRIDE + c + 16] = (_Float16)0.f;  // K-pad zeros for 16-wide layers
    }
  }
}

// ---------------- kernels ----------------
__global__ __launch_bounds__(256)
void prep_weights(const float* wc, const float* bc, const float* win, const float* bin,
                  const float* w4p, const float* b4p, const float* w64p, const float* b64p,
                  const float* w26p, const float* b26p, const float* w32p, const float* b32p,
                  const float* w36p, const float* b36p, const float* w16p, const float* b16p,
                  const float* w47p, const float* b47p, _Float16* wsW, float* wsB) {
  const int li = blockIdx.x;
  const int tid = threadIdx.x;
  LD Ld = g_tab.l[li];

  if (Ld.grp == 8) {  // final layer: keep fp32
    if (tid < 16) wsB[Ld.boff + tid] = w47p[tid];
    if (tid == 16) wsB[Ld.boff + 16] = b47p[0];
    return;
  }

  const float* wsrc; const float* bsrc;
  switch (Ld.grp) {
    case 0:  wsrc = wc;                       bsrc = bc;                break;
    case 1:  wsrc = win + Ld.idx * 256 * 256; bsrc = bin + Ld.idx*256;  break;
    case 2:  wsrc = w4p;                      bsrc = b4p;               break;
    case 3:  wsrc = w64p + Ld.idx * 64 * 64;  bsrc = b64p + Ld.idx*64;  break;
    case 4:  wsrc = w26p;                     bsrc = b26p;              break;
    case 5:  wsrc = w32p + Ld.idx * 32 * 32;  bsrc = b32p + Ld.idx*32;  break;
    case 6:  wsrc = w36p;                     bsrc = b36p;              break;
    default: wsrc = w16p + Ld.idx * 16 * 16;  bsrc = b16p + Ld.idx*16;  break;
  }

  const int N = Ld.N, K = Ld.K, KP = Ld.KP;
  _Float16* dst = wsW + Ld.woff;
  for (int e = tid; e < N * KP; e += 256) {
    int n = e / KP, k = e - n * KP;
    float v = 0.f;
    if (k < K) {
      v = wsrc[n * K + k];
      if (li == 0 && fabsf(v) < 0.01f) v = 0.f;   // CustomLinear hard threshold
    }
    dst[e] = (_Float16)v;
  }
  for (int e = tid; e < N; e += 256) wsB[Ld.boff + e] = bsrc[e];
}

__global__ __launch_bounds__(NTHREADS)
void mlp_fused(const float* __restrict__ x, const _Float16* __restrict__ wsW,
               const float* __restrict__ wsB, float* __restrict__ out) {
  extern __shared__ char smem[];
  _Float16* act   = (_Float16*)smem;                 // 256 x 264 f16
  _Float16* wbig  = act + MTILE * ASTRIDE;           // 256 x 264 f16 (K=256 layers)
  _Float16* wsm0  = wbig + 256 * 264;                // 64 x 72 f16 (double-buffered)
  _Float16* wsm1  = wsm0 + 64 * 72;
  float* bigbias  = (float*)(wsm1 + 64 * 72);        // 256 f32
  float* bbuf     = bigbias + 256;                   // 2 x 64 f32 (double-buffered)
  _Float16* wsmall[2] = {wsm0, wsm1};

  const int tid  = threadIdx.x;
  const int lane = tid & 31;
  const int wave = tid >> 5;
  const int rbase = wave * 16;
  const long long row0 = (long long)blockIdx.x * MTILE;

  // stage x tile: fp32 global -> f16 LDS (streaming, non-temporal)
  const float* xg = x + row0 * 256;
  for (int e = tid; e < MTILE * 64; e += NTHREADS) {   // float4 granules
    int r = e >> 6, c = (e & 63) << 2;
    f32x4 v = __builtin_nontemporal_load((const f32x4*)(xg + r * 256 + c));
    _Float16* d = act + r * ASTRIDE + c;
    d[0] = (_Float16)v[0]; d[1] = (_Float16)v[1];
    d[2] = (_Float16)v[2]; d[3] = (_Float16)v[3];
  }

  // ---- big layers 0..4 (K=256), serial staging through the big slot ----
  for (int li = 0; li <= 4; ++li) {
    const LD Ld = g_tab.l[li];
    __syncthreads();                                  // prior reads of wbig complete
    stage_weights(wbig, wsW + Ld.woff, Ld.KP, Ld.N, tid);
    if (tid < Ld.N) bigbias[tid] = wsB[Ld.boff + tid];
    wait_tensorcnt0();
    __syncthreads();
    if (li == 4) {                                    // prefetch first small layer
      const LD L5 = g_tab.l[5];
      stage_weights(wsmall[0], wsW + L5.woff, L5.KP, L5.N, tid);
      if (tid < L5.N) bbuf[tid] = wsB[L5.boff + tid];
    }
    if (li < 4) layer_compute<16, 256, 264>(act, wbig, bigbias, lane, rbase, false);
    else        layer_compute<4,  256, 264>(act, wbig, bigbias, lane, rbase, false);
  }

  // ---- small layers 5..46: double-buffered TDM prefetch of layer li+1 ----
  int cur = 0;
  for (int li = 5; li <= 46; ++li) {
    wait_tensorcnt0();                                // TDM for layer li complete
    __syncthreads();                                  // all waves done with layer li-1 buffers
    if (li < 46) {
      const LD Ln = g_tab.l[li + 1];
      stage_weights(wsmall[cur ^ 1], wsW + Ln.woff, Ln.KP, Ln.N, tid);
      if (tid < Ln.N) bbuf[(cur ^ 1) * 64 + tid] = wsB[Ln.boff + tid];
    }
    const _Float16* W = wsmall[cur];
    const float* Bv = bbuf + cur * 64;
    if      (li <= 25) layer_compute<4, 64, 72>(act, W, Bv, lane, rbase, false);
    else if (li == 26) layer_compute<2, 64, 72>(act, W, Bv, lane, rbase, false);
    else if (li <= 35) layer_compute<2, 32, 40>(act, W, Bv, lane, rbase, false);
    else               layer_compute<1, 32, 40>(act, W, Bv, lane, rbase, true);
    cur ^= 1;
  }

  // ---- final layer fc47: 16 -> 1, fp32 scalar dot ----
  const float* w47 = wsB + g_tab.l[47].boff;
  const float  b47 = w47[16];
  if (lane < 16) {
    int row = rbase + lane;
    const _Float16* ar = act + row * ASTRIDE;
    float s = b47;
#pragma unroll
    for (int k = 0; k < 16; ++k) s += (float)ar[k] * w47[k];
    __builtin_nontemporal_store(s, out + row0 + row);
  }
}

// ---------------- launch ----------------
extern "C" void kernel_launch(void* const* d_in, const int* in_sizes, int n_in,
                              void* d_out, int out_size, void* d_ws, size_t ws_size,
                              hipStream_t stream) {
  (void)n_in; (void)out_size; (void)ws_size;
  const float* x    = (const float*)d_in[0];
  const float* wc   = (const float*)d_in[1];
  const float* bc   = (const float*)d_in[2];
  const float* win  = (const float*)d_in[3];
  const float* bin  = (const float*)d_in[4];
  const float* w4p  = (const float*)d_in[5];
  const float* b4p  = (const float*)d_in[6];
  const float* w64p = (const float*)d_in[7];
  const float* b64p = (const float*)d_in[8];
  const float* w26p = (const float*)d_in[9];
  const float* b26p = (const float*)d_in[10];
  const float* w32p = (const float*)d_in[11];
  const float* b32p = (const float*)d_in[12];
  const float* w36p = (const float*)d_in[13];
  const float* b36p = (const float*)d_in[14];
  const float* w16p = (const float*)d_in[15];
  const float* b16p = (const float*)d_in[16];
  const float* w47p = (const float*)d_in[17];
  const float* b47p = (const float*)d_in[18];

  _Float16* wsW = (_Float16*)d_ws;
  float*    wsB = (float*)((char*)d_ws + (size_t)WTOT_HALVES * 2);

  prep_weights<<<48, 256, 0, stream>>>(wc, bc, win, bin, w4p, b4p, w64p, b64p,
                                       w26p, b26p, w32p, b32p, w36p, b36p,
                                       w16p, b16p, w47p, b47p, wsW, wsB);

  const int B = in_sizes[0] / 256;                        // 131072
  const size_t shmem =
      ((size_t)MTILE * ASTRIDE + 256 * 264 + 2 * 64 * 72) * 2   // act + big W + 2 small W
      + (256 + 128) * 4;                                        // bigbias + 2x64 bias
  mlp_fused<<<B / MTILE, NTHREADS, shmem, stream>>>(x, wsW, wsB, (float*)d_out);
}
